// Transducer_1357209666390
// MI455X (gfx1250) — matrix operand order repaired and checked by reading, not compiled, not run on hardware
//
#include <hip/hip_runtime.h>
#include <hip/hip_bf16.h>

// ---------------------------------------------------------------------------
// RNN-Transducer for MI455X (gfx1250, wave32, WMMA bf16 16x16x32, f32 accum)
// B=4 T=128 U=64 FEAT=80 H=1024 VOCAB=64 EMB=64 JOINT=640 SMAX=4096
//
// LSTMs: xg = X@W+b via WMMA GEMM (A tile fully staged in LDS once); per-step
//        kernel computes h@U (batch padded 4->16) with 4 gate tiles per wave.
// Joint: fused 3-GEMM per 32-row tile with softmax folded into the last GEMM
//        via a ones-column appended to Wp (row-sum rides the WMMA accumulator).
// Wave-uniform tile indices go through readfirstlane so tile offsets live in
// SGPRs; constant tile strides fold into load immediate offsets.
// ---------------------------------------------------------------------------

typedef __attribute__((ext_vector_type(16))) __bf16 v16bf;
typedef __attribute__((ext_vector_type(8)))  float  v8f;

#define DEV __device__ __forceinline__

DEV v8f wmma_bf16(v16bf a, v16bf b, v8f c) {
  return __builtin_amdgcn_wmma_f32_16x16x32_bf16(false, a, false, b,
                                                 (short)0, c, false, false);
}

// A-operand LDS slot for element (m, k) of a 16x32 bf16 tile (ISA 7.12.2):
// lanes 0-15 hold K={0..7,16..23}, lanes 16-31 hold K={8..15,24..31}.
// Even/odd k land in adjacent slots -> pairs pack into one u32 store.
DEV int a_slot(int m, int k) {
  int hi = k >> 4, k4 = k & 15, g = k4 >> 3, k8 = k4 & 7;
  int e = (((hi << 2) + (k8 >> 1)) << 1) + (k8 & 1);
  return ((m + (g << 4)) << 4) + e;
}

DEV unsigned pack_bf16x2(float lo, float hi) {
  unsigned short a = __builtin_bit_cast(unsigned short, (__bf16)lo);
  unsigned short b = __builtin_bit_cast(unsigned short, (__bf16)hi);
  return (unsigned)a | ((unsigned)b << 16);
}

DEV float sigf(float x)   { return 1.0f / (1.0f + __expf(-x)); }
DEV float tanhf_(float x) { return 2.0f / (1.0f + __expf(-2.0f * x)) - 1.0f; }

// ---------------------------------------------------------------------------
// Pack row-major f32 weight [K, Nreal] into bf16 WMMA-B tiles.
// B layout (32x16): lane = n + 16*((klocal/8)%2), element e = 8*(klocal/16) + klocal%8.
// Tile (nt,kt) at out[(nt*ktiles+kt)*512]; cols n>=Nreal are 0 except ones_col.
// ---------------------------------------------------------------------------
__global__ __launch_bounds__(32)
void pack_b_kernel(const float* __restrict__ W, int K, int Nreal, int ones_col,
                   __bf16* __restrict__ out, int ktiles) {
  int tile = blockIdx.x;
  int kt = tile % ktiles, nt = tile / ktiles;
  int lane = threadIdx.x & 31;
  int g = lane >> 4, nl = lane & 15;
  int n = nt * 16 + nl;
  __bf16* o = out + ((size_t)(nt * ktiles + kt) << 9) + lane * 16;
#pragma unroll
  for (int e = 0; e < 16; ++e) {
    int k = kt * 32 + ((e >> 3) << 4) + (g << 3) + (e & 7);
    float v = 0.0f;
    if (k < K) {
      if (n < Nreal)          v = W[(size_t)k * Nreal + n];
      else if (n == ones_col) v = 1.0f;
    }
    o[e] = (__bf16)v;
  }
}

// ---------------------------------------------------------------------------
// C[M,N] = A[M,K](f32) @ Bpacked + bias. Whole 16xK A tile staged once
// (K <= 1024 -> <= 32KB). grid = (Ntiles/8, M/16), 256 threads.
// ---------------------------------------------------------------------------
__global__ __launch_bounds__(256)
void gemm_bias_kernel(const float* __restrict__ A, int lda, int M,
                      const __bf16* __restrict__ Bp, int ktiles,
                      const float* __restrict__ bias,
                      float* __restrict__ C, int ldc, int K) {
  __shared__ __align__(32) __bf16 As[32 * 512];
  int tid = threadIdx.x, lane = tid & 31;
  int mt = blockIdx.y;
  int nt = __builtin_amdgcn_readfirstlane(blockIdx.x * 8 + (tid >> 5));
  // stage: s -> (kt, m, kpair); branchless clamped float2 loads, packed stores
  for (int s = tid; s < ktiles * 256; s += 256) {
    int kt = s >> 8, m = (s >> 4) & 15, kp = s & 15;
    int k = kt * 32 + kp * 2;
    int gm = mt * 16 + m;
    bool ok = (gm < M) && (k < K);          // K even -> pair-granular validity
    const float* ap = A + (size_t)(ok ? gm : 0) * lda + (ok ? k : 0);
    float2 v = *(const float2*)ap;
    ((unsigned*)As)[(kt * 512 + a_slot(m, k & 31)) >> 1] =
        pack_bf16x2(ok ? v.x : 0.0f, ok ? v.y : 0.0f);
  }
  __syncthreads();
  const __bf16* bpl = Bp + ((size_t)nt * ktiles << 9) + lane * 16;   // saddr form
  v8f acc = {};
  for (int kt = 0; kt < ktiles; ++kt) {
    v16bf a  = *(const v16bf*)(As + kt * 512 + lane * 16);
    v16bf bb = *(const v16bf*)(bpl + (kt << 9));
    acc = wmma_bf16(a, bb, acc);
  }
  int col = nt * 16 + (lane & 15);
  float bv = bias ? bias[col] : 0.0f;
  int mbase = (lane >> 4) << 3;
#pragma unroll
  for (int i = 0; i < 8; ++i) {
    int m = mt * 16 + mbase + i;
    if (m < M) C[(size_t)m * ldc + col] = acc[i] + bv;
  }
}

// ---------------------------------------------------------------------------
// One LSTM step: gates = xg[:,t,:] + h @ U; update h,c; emit hseq[:,t,:].
// grid = (8,1), 256 threads: wave owns 16 H cols x 4 gates.
// Gate tiles are +2MB apart in the packed layout -> immediate offsets.
// ---------------------------------------------------------------------------
__global__ __launch_bounds__(256)
void lstm_step_kernel(const float* __restrict__ xg,   // [4, Tdim, 4096]
                      const float* __restrict__ hin,  // [4,1024]
                      float* __restrict__ hout,       // [4,1024]
                      float* __restrict__ cbuf,       // [4,1024] in/out
                      const __bf16* __restrict__ Up,  // ktiles=32, ntiles=256
                      float* __restrict__ hseq,       // [4, Tdim, 1024]
                      int t, int Tdim) {
  __shared__ __align__(32) __bf16 hA[32 * 512];       // padded h, A layout
  int tid = threadIdx.x, lane = tid & 31;
  for (int s = tid; s < 16 * 512; s += 256) {         // (m, kpair)
    int m = s >> 9, k = (s & 511) * 2;
    bool ok = m < 4;
    const float* ap = hin + (size_t)(ok ? m : 0) * 1024 + k;
    float2 v = *(const float2*)ap;
    ((unsigned*)hA)[((k >> 5) * 512 + a_slot(m, k & 31)) >> 1] =
        pack_bf16x2(ok ? v.x : 0.0f, ok ? v.y : 0.0f);
  }
  __syncthreads();
  int n0t = __builtin_amdgcn_readfirstlane(blockIdx.x * 8 + (tid >> 5));
  const __bf16* upl = Up + ((size_t)n0t * 32 << 9) + lane * 16;
  const int GATE = 64 * 32 * 512;                     // elems between gates
  v8f ai = {}, af = {}, ag = {}, ao = {};
  for (int kt = 0; kt < 32; ++kt) {
    v16bf a  = *(const v16bf*)(hA + kt * 512 + lane * 16);
    const __bf16* bt = upl + (kt << 9);
    v16bf b0 = *(const v16bf*)(bt);
    v16bf b1 = *(const v16bf*)(bt + GATE);
    v16bf b2 = *(const v16bf*)(bt + 2 * GATE);
    v16bf b3 = *(const v16bf*)(bt + 3 * GATE);
    ai = wmma_bf16(a, b0, ai);
    af = wmma_bf16(a, b1, af);
    ag = wmma_bf16(a, b2, ag);
    ao = wmma_bf16(a, b3, ao);
  }
  if (lane < 16) {
    int col = n0t * 16 + lane;
#pragma unroll
    for (int m = 0; m < 4; ++m) {
      const float* xgr = xg + ((size_t)m * Tdim + t) * 4096;
      float ig = ai[m] + xgr[col];
      float fg = af[m] + xgr[1024 + col];
      float gg = ag[m] + xgr[2048 + col];
      float og = ao[m] + xgr[3072 + col];
      float cn = sigf(fg) * cbuf[m * 1024 + col] + sigf(ig) * tanhf_(gg);
      float hn = sigf(og) * tanhf_(cn);
      cbuf[m * 1024 + col] = cn;
      hout[m * 1024 + col] = hn;
      hseq[((size_t)m * Tdim + t) * 1024 + col] = hn;
    }
  }
}

// ---------------------------------------------------------------------------
// Fused joint per 32-row tile (2 row groups of 16) for 2x weight reuse.
// grid = (1024), 256 threads.
// ---------------------------------------------------------------------------
__global__ __launch_bounds__(256)
void joint_kernel(const float* __restrict__ x2,   // [4,128,1024]
                  const float* __restrict__ p2,   // [4, 64,1024]
                  const __bf16* __restrict__ Wjp, // ktiles=32, ntiles=40
                  const float* __restrict__ bj,
                  const __bf16* __restrict__ Wsp, // ktiles=20, ntiles=256
                  const float* __restrict__ bs,
                  const __bf16* __restrict__ Wpp, // ktiles=128, ntiles=5 (col64=1)
                  const float* __restrict__ bp,
                  float* __restrict__ out) {      // [32768, 64]
  __shared__ __align__(32) __bf16 jdA[2][20 * 512]; // 40KB: jd, A layout
  __shared__ __align__(32) __bf16 jAh[16 * 512];    // 16KB: half-K j staging,
                                                    // reused as E scratch, then red
  __bf16* eS  = jAh;                                // 8 waves x 1024 bf16
  float*  red = (float*)jAh;                        // 32x80 f32 (10KB)

  int tid = threadIdx.x, lane = tid & 31;
  int wq = __builtin_amdgcn_readfirstlane(tid >> 5);  // wave id, scalar
  int g = lane >> 4, nl = lane & 15;
  int r0 = blockIdx.x * 32;
  int b  = r0 >> 13;
  int tt = (r0 >> 6) & 127;
  int u0 = r0 & 63;                                  // 0 or 32
  const float* xrow = x2 + ((size_t)b * 128 + tt) * 1024;
  const float* prow = p2 + ((size_t)b * 64 + u0) * 1024;

  // ---- GEMM1: jd[32,640] = (x+p) @ Wj + bj, K staged in two 512 halves ----
  const __bf16* wjl = Wjp + ((size_t)(wq * 5) * 32 << 9) + lane * 16;
  for (int rg = 0; rg < 2; ++rg) {
    v8f acc[5] = {};
    for (int kh = 0; kh < 2; ++kh) {
      __syncthreads();
      for (int s = tid; s < 16 * 256; s += 256) {    // (m, kpair) in half
        int m = s >> 8, k = kh * 512 + (s & 255) * 2;
        const float* pr = prow + (size_t)(rg * 16 + m) * 1024 + k;
        float2 xv = *(const float2*)(xrow + k);
        float2 pv = *(const float2*)pr;
        int kloc = k - kh * 512;
        ((unsigned*)jAh)[((kloc >> 5) * 512 + a_slot(m, k & 31)) >> 1] =
            pack_bf16x2(xv.x + pv.x, xv.y + pv.y);
      }
      __syncthreads();
      for (int kt = 0; kt < 16; ++kt) {
        v16bf a = *(const v16bf*)(jAh + kt * 512 + lane * 16);
        const __bf16* bt = wjl + ((kh * 16 + kt) << 9);
#pragma unroll
        for (int j = 0; j < 5; ++j) {                // +32KB per n-tile
          v16bf bb = *(const v16bf*)(bt + j * (32 * 512));
          acc[j] = wmma_bf16(a, bb, acc[j]);
        }
      }
    }
#pragma unroll
    for (int j = 0; j < 5; ++j) {
      int c = (wq * 5 + j) * 16 + nl;
      float bv = bj[c];
#pragma unroll
      for (int i = 0; i < 8; ++i)
        jdA[rg][(c >> 5) * 512 + a_slot(i + (g << 3), c & 31)] =
            (__bf16)(acc[j][i] + bv);
    }
  }
  __syncthreads();

  // ---- GEMM2 + exp + GEMM3 (rowsum via Wp ones column) --------------------
  v8f oacc0[5] = {}, oacc1[5] = {};
  __bf16* eMy = eS + wq * 1024;                      // two 16x32 E tiles
  const __bf16* wpl = Wpp + lane * 16;
  for (int ch = wq; ch < 128; ch += 8) {
    if (ch + 8 < 128)                                // gfx1250 prefetch path
      __builtin_prefetch(Wsp + ((size_t)(2 * (ch + 8) * 20) << 9), 0, 0);
    const __bf16* wsl = Wsp + ((size_t)(2 * ch * 20) << 9) + lane * 16;
    v8f l00 = {}, l01 = {}, l10 = {}, l11 = {};
    for (int kt = 0; kt < 20; ++kt) {
      v16bf a0 = *(const v16bf*)(jdA[0] + kt * 512 + lane * 16);
      v16bf a1 = *(const v16bf*)(jdA[1] + kt * 512 + lane * 16);
      const __bf16* bt = wsl + (kt << 9);
      v16bf b0 = *(const v16bf*)(bt);                // n-tile 2ch
      v16bf b1 = *(const v16bf*)(bt + 20 * 512);     // n-tile 2ch+1 (+20KB)
      l00 = wmma_bf16(a0, b0, l00);
      l01 = wmma_bf16(a0, b1, l01);
      l10 = wmma_bf16(a1, b0, l10);
      l11 = wmma_bf16(a1, b1, l11);
    }
    float bs0 = bs[ch * 32 + nl], bs1 = bs[ch * 32 + 16 + nl];
#pragma unroll
    for (int i = 0; i < 8; ++i) {
      int m = i + (g << 3);
      eMy[a_slot(m, nl)]            = (__bf16)__expf(l00[i] + bs0);
      eMy[a_slot(m, 16 + nl)]       = (__bf16)__expf(l01[i] + bs1);
      eMy[512 + a_slot(m, nl)]      = (__bf16)__expf(l10[i] + bs0);
      eMy[512 + a_slot(m, 16 + nl)] = (__bf16)__expf(l11[i] + bs1);
    }
    asm volatile("s_wait_dscnt 0" ::: "memory");     // cross-lane LDS RAW
    v16bf ea0 = *(const v16bf*)(eMy + lane * 16);
    v16bf ea1 = *(const v16bf*)(eMy + 512 + lane * 16);
    const __bf16* wpt = wpl + ((size_t)ch << 9);
#pragma unroll
    for (int j = 0; j < 5; ++j) {                    // +128KB per n-tile
      v16bf bb = *(const v16bf*)(wpt + (size_t)j * (128 * 512));
      oacc0[j] = wmma_bf16(ea0, bb, oacc0[j]);
      oacc1[j] = wmma_bf16(ea1, bb, oacc1[j]);
    }
  }

  // ---- cross-wave reduction (red overlays eS after barrier) ---------------
  __syncthreads();
  for (int s = tid; s < 32 * 80; s += 256) red[s] = 0.0f;
  __syncthreads();
#pragma unroll
  for (int j = 0; j < 5; ++j)
#pragma unroll
    for (int i = 0; i < 8; ++i) {
      int m = i + (g << 3), n = j * 16 + nl;
      atomicAdd(&red[m * 80 + n],        oacc0[j][i]);
      atomicAdd(&red[(16 + m) * 80 + n], oacc1[j][i]);
    }
  __syncthreads();

  // ---- out = (E@Wp)/rowsum + bp -------------------------------------------
  for (int s = tid; s < 32 * 64; s += 256) {
    int m = s >> 6, n = s & 63;
    out[(size_t)(r0 + m) * 64 + n] = red[m * 80 + n] / red[m * 80 + 64] + bp[n];
  }
}

// ---------------------------------------------------------------------------
__global__ __launch_bounds__(256)
void embed_kernel(const int* __restrict__ idx, const float* __restrict__ emb,
                  float* __restrict__ o) {
  int t = blockIdx.x * 256 + threadIdx.x;
  if (t < 256 * 64) o[t] = emb[idx[t >> 6] * 64 + (t & 63)];
}

__global__ __launch_bounds__(256)
void copy_f32(float* __restrict__ d, const float* __restrict__ s, int n) {
  int i = blockIdx.x * 256 + threadIdx.x;
  if (i < n) d[i] = s[i];
}

__global__ __launch_bounds__(256)
void zero_f32(float* __restrict__ d, int n) {
  int i = blockIdx.x * 256 + threadIdx.x;
  if (i < n) d[i] = 0.0f;
}

// ---------------------------------------------------------------------------
extern "C" void kernel_launch(void* const* d_in, const int* in_sizes, int n_in,
                              void* d_out, int out_size, void* d_ws, size_t ws_size,
                              hipStream_t stream) {
  (void)in_sizes; (void)n_in; (void)out_size; (void)ws_size;
  const float* encoder_inp = (const float*)d_in[0];
  const int*   pred_inp    = (const int*)d_in[1];
  const float* enc_state   = (const float*)d_in[2];
  const float* enc_W0 = (const float*)d_in[3];
  const float* enc_U0 = (const float*)d_in[4];
  const float* enc_b0 = (const float*)d_in[5];
  const float* enc_W1 = (const float*)d_in[6];
  const float* enc_U1 = (const float*)d_in[7];
  const float* enc_b1 = (const float*)d_in[8];
  const float* embedW = (const float*)d_in[9];
  const float* pred_W0 = (const float*)d_in[10];
  const float* pred_U0 = (const float*)d_in[11];
  const float* pred_b0 = (const float*)d_in[12];
  const float* pred_W1 = (const float*)d_in[13];
  const float* pred_U1 = (const float*)d_in[14];
  const float* pred_b1 = (const float*)d_in[15];
  const float* Wj = (const float*)d_in[16];
  const float* bj = (const float*)d_in[17];
  const float* Ws = (const float*)d_in[18];
  const float* bs = (const float*)d_in[19];
  const float* Wp = (const float*)d_in[20];
  const float* bp = (const float*)d_in[21];
  float* out = (float*)d_out;

  char* ws = (char*)d_ws;
  size_t off = 0;
  auto alloc = [&](size_t bytes) -> void* {
    void* p = ws + off;
    off = (off + bytes + 255) & ~(size_t)255;
    return p;
  };
  auto pack = [&](const float* W, int K, int Nreal, int ntiles, int ones_col) -> __bf16* {
    int ktiles = (K + 31) >> 5;
    __bf16* p = (__bf16*)alloc((size_t)ktiles * ntiles * 512 * sizeof(__bf16));
    pack_b_kernel<<<dim3(ktiles * ntiles), dim3(32), 0, stream>>>(
        W, K, Nreal, ones_col, p, ktiles);
    return p;
  };

  __bf16* encW0p = pack(enc_W0,   80, 4096, 256, -1);
  __bf16* encU0p = pack(enc_U0, 1024, 4096, 256, -1);
  __bf16* encW1p = pack(enc_W1, 1024, 4096, 256, -1);
  __bf16* encU1p = pack(enc_U1, 1024, 4096, 256, -1);
  __bf16* prdW0p = pack(pred_W0,  64, 4096, 256, -1);
  __bf16* prdU0p = pack(pred_U0, 1024, 4096, 256, -1);
  __bf16* prdW1p = pack(pred_W1, 1024, 4096, 256, -1);
  __bf16* prdU1p = pack(pred_U1, 1024, 4096, 256, -1);
  __bf16* Wjp = pack(Wj, 1024,  640,  40, -1);
  __bf16* Wsp = pack(Ws,  640, 4096, 256, -1);
  __bf16* Wpp = pack(Wp, 4096,   64,   5, 64);     // col 64 = ones (rowsum)

  float* xg   = (float*)alloc((size_t)512 * 4096 * 4);
  float* x1   = (float*)alloc((size_t)512 * 1024 * 4);
  float* x2   = (float*)alloc((size_t)512 * 1024 * 4);
  float* pemb = (float*)alloc((size_t)256 * 64 * 4);
  float* p1   = (float*)alloc((size_t)256 * 1024 * 4);
  float* p2   = (float*)alloc((size_t)256 * 1024 * 4);
  float* hb0  = (float*)alloc(4096 * 4);
  float* hb1  = (float*)alloc(4096 * 4);
  float* cb   = (float*)alloc(4096 * 4);

  auto gemm = [&](const float* A, int lda, int M, int K, const __bf16* Bp,
                  int Ntiles, const float* bias, float* C, int ldc) {
    int ktiles = (K + 31) >> 5;
    gemm_bias_kernel<<<dim3(Ntiles / 8, (M + 15) / 16), dim3(256), 0, stream>>>(
        A, lda, M, Bp, ktiles, bias, C, ldc, K);
  };

  // ---- encoder layer 0 ----------------------------------------------------
  gemm(encoder_inp, 80, 512, 80, encW0p, 256, enc_b0, xg, 4096);
  copy_f32<<<16, 256, 0, stream>>>(hb0, enc_state, 4096);
  copy_f32<<<16, 256, 0, stream>>>(cb, enc_state + 4096, 4096);
  for (int t = 0; t < 128; ++t) {
    float* hin  = (t & 1) ? hb1 : hb0;
    float* hout = (t & 1) ? hb0 : hb1;
    lstm_step_kernel<<<8, 256, 0, stream>>>(xg, hin, hout, cb, encU0p, x1, t, 128);
  }
  // ---- encoder layer 1 (same incoming state, per reference) ---------------
  gemm(x1, 1024, 512, 1024, encW1p, 256, enc_b1, xg, 4096);
  copy_f32<<<16, 256, 0, stream>>>(hb0, enc_state, 4096);
  copy_f32<<<16, 256, 0, stream>>>(cb, enc_state + 4096, 4096);
  for (int t = 0; t < 128; ++t) {
    float* hin  = (t & 1) ? hb1 : hb0;
    float* hout = (t & 1) ? hb0 : hb1;
    lstm_step_kernel<<<8, 256, 0, stream>>>(xg, hin, hout, cb, encU1p, x2, t, 128);
  }
  copy_f32<<<16, 256, 0, stream>>>(out + 2097152, hb0, 4096);          // h (t=127 -> hb0)
  copy_f32<<<16, 256, 0, stream>>>(out + 2097152 + 4096, cb, 4096);    // c

  // ---- prediction network (zero initial state) ----------------------------
  embed_kernel<<<64, 256, 0, stream>>>(pred_inp, embedW, pemb);
  gemm(pemb, 64, 256, 64, prdW0p, 256, pred_b0, xg, 4096);
  zero_f32<<<16, 256, 0, stream>>>(hb0, 4096);
  zero_f32<<<16, 256, 0, stream>>>(cb, 4096);
  for (int t = 0; t < 64; ++t) {
    float* hin  = (t & 1) ? hb1 : hb0;
    float* hout = (t & 1) ? hb0 : hb1;
    lstm_step_kernel<<<8, 256, 0, stream>>>(xg, hin, hout, cb, prdU0p, p1, t, 64);
  }
  gemm(p1, 1024, 256, 1024, prdW1p, 256, pred_b1, xg, 4096);
  zero_f32<<<16, 256, 0, stream>>>(hb0, 4096);
  zero_f32<<<16, 256, 0, stream>>>(cb, 4096);
  for (int t = 0; t < 64; ++t) {
    float* hin  = (t & 1) ? hb1 : hb0;
    float* hout = (t & 1) ? hb0 : hb1;
    lstm_step_kernel<<<8, 256, 0, stream>>>(xg, hin, hout, cb, prdU1p, p2, t, 64);
  }

  // ---- fused joint + softmax + projection ---------------------------------
  joint_kernel<<<1024, 256, 0, stream>>>(x2, p2, Wjp, bj, Wsp, bs, Wpp, bp, out);
}
